// MultiAxisAttention_55448027791945
// MI455X (gfx1250) — matrix operand order repaired and checked by reading, not compiled, not run on hardware
//
#include <hip/hip_runtime.h>

// ---------------------------------------------------------------------------
// MultiAxisAttention for MI455X (gfx1250, wave32, WMMA, double-buffered
// async-to-LDS staging).
// Pipeline (bf16 intermediates in d_ws, f32 accumulation in WMMA):
//   0) pre-transpose weights to bf16 [n][k] copies (resident in L2)
//   1) Q = queries*Wq+bq ; K = values*Wk+bk ; V = values*Wv+bv   (bf16 out)
//   2) axis-H attention on heads 0..5, axis-W attention on heads 6..11
//   3) out = O*Wo + bo  (f32 out)
// Workspace: 4 * 65536*768 bf16 matrices (402.7 MB) + 4 * 768*768 bf16 weights.
// ---------------------------------------------------------------------------

typedef __attribute__((ext_vector_type(16))) __bf16 v16bf;
typedef __attribute__((ext_vector_type(8)))  __bf16 v8bf;
typedef __attribute__((ext_vector_type(4)))  __bf16 v4bf;
typedef __attribute__((ext_vector_type(8)))  float  v8f;

#define MTOT 65536
#define KDIM 768
#define NDIM 768

// --- CDNA5 async copy: global -> LDS, 16 bytes per lane, ASYNCcnt-tracked ---
static __device__ __forceinline__ unsigned lds_off32(const void* p) {
  // low 32 bits of a generic shared-aperture address == LDS byte offset
  return (unsigned)(unsigned long long)p;
}
static __device__ __forceinline__ void async_copy_b128(unsigned lds_off, const void* gptr) {
  asm volatile("global_load_async_to_lds_b128 %0, %1, off"
               :: "v"(lds_off), "v"(gptr) : "memory");
}
template<int N>
static __device__ __forceinline__ void async_wait_le() {
  asm volatile("s_wait_asynccnt %0" :: "i"(N) : "memory");
}

// A-operand fragment (16x32 bf16): lane holds row, lane-half lh selects
// K chunks {0..7,16..23} (lh=0) or {8..15,24..31} (lh=1).
static __device__ __forceinline__ v16bf frag_a(const __bf16* row32, int lh) {
  v8bf lo = *(const v8bf*)(row32 + lh * 8);
  v8bf hi = *(const v8bf*)(row32 + 16 + lh * 8);
  return __builtin_shufflevector(lo, hi, 0,1,2,3,4,5,6,7,8,9,10,11,12,13,14,15);
}
// B-operand fragment (32x16 bf16): lane holds column n, lane-half lh selects
// K chunks {0..7,16..23} (lh=0) or {8..15,24..31} (lh=1) (n-major staging).
static __device__ __forceinline__ v16bf frag_b(const __bf16* col32, int lh) {
  v8bf lo = *(const v8bf*)(col32 + lh * 8);
  v8bf hi = *(const v8bf*)(col32 + 16 + lh * 8);
  return __builtin_shufflevector(lo, hi, 0,1,2,3,4,5,6,7,8,9,10,11,12,13,14,15);
}

// ---------------------------------------------------------------------------
// Weight pre-pass: W[k][n] f32 row-major -> Wt[n][k] bf16 (one-time, L2-hot)
// ---------------------------------------------------------------------------
__global__ __launch_bounds__(256)
void transpose_w_bf16(const float* __restrict__ W, __bf16* __restrict__ Wt) {
  const int k = blockIdx.y;
  const int n = blockIdx.x * 256 + threadIdx.x;
  Wt[(size_t)n * KDIM + k] = (__bf16)W[(size_t)k * NDIM + n];
}

// ---------------------------------------------------------------------------
// Dense GEMM  C[M,768] = A[M,768] * Bt^T + bias, block tile 128x128,
// 8 waves -> each wave 64(M) x 32(N) = 4x2 WMMA tiles of 16x16, f32 acc.
// Double-buffered pipeline: tile i+1 is prefetched (async-to-LDS for bf16
// operands, global->reg for the fp32 A path) while tile i runs on WMMA;
// s_wait_asynccnt <=N only drains the older tile's copies.
// ---------------------------------------------------------------------------
template<bool IN_BF16, bool OUT_F32>
__global__ __launch_bounds__(256)
void fused_proj_gemm(const void* __restrict__ Av, const __bf16* __restrict__ Bt,
                     const float* __restrict__ bias, void* __restrict__ Cv) {
  __shared__ __bf16 As[2][128 * 32];     // [m][k] row-major
  __shared__ __bf16 Bs[2][128 * 32];     // [n][k]

  const int tid  = threadIdx.x;
  const int m0   = blockIdx.y * 128;
  const int n1   = blockIdx.x * 128;
  const int wv   = tid >> 5, lane = tid & 31;
  const int lh   = lane >> 4, lrow = lane & 15;
  const int wm   = wv & 1, wn = wv >> 1;
  const int rT   = tid >> 1, cT = (tid & 1) * 16;  // 16 elems (32 B) per thread

  const __bf16* gA16 = (const __bf16*)Av + (size_t)(m0 + rT) * KDIM + cT;
  const float*  gA32 = (const float*)Av  + (size_t)(m0 + rT) * KDIM + cT;
  const __bf16* gB   = Bt + (size_t)(n1 + rT) * KDIM + cT;

  v8f acc[4][2] = {};
  float4 fa[4];

  // ---- prologue: start tile 0 ----
  if constexpr (IN_BF16) {
    async_copy_b128(lds_off32(&As[0][rT * 32 + cT]),     gA16);
    async_copy_b128(lds_off32(&As[0][rT * 32 + cT + 8]), gA16 + 8);
  } else {
#pragma unroll
    for (int j = 0; j < 4; ++j) fa[j] = *(const float4*)(gA32 + j * 4);
  }
  async_copy_b128(lds_off32(&Bs[0][rT * 32 + cT]),     gB);
  async_copy_b128(lds_off32(&Bs[0][rT * 32 + cT + 8]), gB + 8);

  for (int i = 0; i < KDIM / 32; ++i) {
    const int p = i & 1;

    // ---- fp32 A path: commit tile-i registers -> LDS (packed cvt, b64) ----
    if constexpr (!IN_BF16) {
#pragma unroll
      for (int j = 0; j < 4; ++j) {
        v4bf q; q[0] = (__bf16)fa[j].x; q[1] = (__bf16)fa[j].y;
                q[2] = (__bf16)fa[j].z; q[3] = (__bf16)fa[j].w;
        *(v4bf*)&As[p][rT * 32 + cT + j * 4] = q;
      }
    }
    // ---- prefetch tile i+1 into the other buffer (stays in flight
    //      through this iteration's WMMA work) ----
    float4 fb[4];
    if (i < KDIM / 32 - 1) {
      const int kn = (i + 1) * 32;
      if constexpr (IN_BF16) {
        async_copy_b128(lds_off32(&As[p ^ 1][rT * 32 + cT]),     gA16 + kn);
        async_copy_b128(lds_off32(&As[p ^ 1][rT * 32 + cT + 8]), gA16 + kn + 8);
      } else {
#pragma unroll
        for (int j = 0; j < 4; ++j) fb[j] = *(const float4*)(gA32 + kn + j * 4);
      }
      async_copy_b128(lds_off32(&Bs[p ^ 1][rT * 32 + cT]),     gB + kn);
      async_copy_b128(lds_off32(&Bs[p ^ 1][rT * 32 + cT + 8]), gB + kn + 8);
      // drain only the older (tile i) copies; tile i+1 remains in flight
      if constexpr (IN_BF16) async_wait_le<4>(); else async_wait_le<2>();
    } else {
      async_wait_le<0>();
    }
    __syncthreads();

    v16bf af[4], bfr[2];
#pragma unroll
    for (int mt = 0; mt < 4; ++mt)
      af[mt] = frag_a(&As[p][(wm * 64 + mt * 16 + lrow) * 32], lh);
#pragma unroll
    for (int nt = 0; nt < 2; ++nt)
      bfr[nt] = frag_b(&Bs[p][(wn * 32 + nt * 16 + lrow) * 32], lh);
#pragma unroll
    for (int mt = 0; mt < 4; ++mt)
#pragma unroll
      for (int nt = 0; nt < 2; ++nt)
        acc[mt][nt] = __builtin_amdgcn_wmma_f32_16x16x32_bf16(
            false, af[mt], false, bfr[nt], (short)0, acc[mt][nt], false, false);
    __syncthreads();   // buffer p free for reuse at iteration i+2

    if constexpr (!IN_BF16) {
#pragma unroll
      for (int j = 0; j < 4; ++j) fa[j] = fb[j];
    }
  }

  // ---- epilogue: C tile layout row = 8*lh + v, col = lrow ----
#pragma unroll
  for (int mt = 0; mt < 4; ++mt) {
#pragma unroll
    for (int nt = 0; nt < 2; ++nt) {
      const int gn  = n1 + wn * 32 + nt * 16 + lrow;
      const float bb = bias[gn];
      const int gm0 = m0 + wm * 64 + mt * 16 + lh * 8;
#pragma unroll
      for (int vv = 0; vv < 8; ++vv) {
        float val = acc[mt][nt][vv] + bb;
        if constexpr (OUT_F32)
          ((float*)Cv)[(size_t)(gm0 + vv) * NDIM + gn] = val;
        else
          ((__bf16*)Cv)[(size_t)(gm0 + vv) * NDIM + gn] = (__bf16)val;
      }
    }
  }
}

// ---------------------------------------------------------------------------
// Axis attention: one block per (head, fixed-coord, batch) 64x64x64 tile.
// AXIS==0: attend along H (row stride 64*768, heads 0..5)
// AXIS==1: attend along W (row stride 768,     heads 6..11)
// 128 threads = 4 waves; wave wv owns query rows wv*16..wv*16+15.
// ---------------------------------------------------------------------------
template<int AXIS>
__global__ __launch_bounds__(128)
void axis_attn(const __bf16* __restrict__ q, const __bf16* __restrict__ k,
               const __bf16* __restrict__ v, __bf16* __restrict__ o) {
  __shared__ __bf16 Qs[64 * 64];      // [row][d]
  __shared__ __bf16 Ks[64 * 64];      // [kpos][d]  (== B^T operand for Q*K^T)
  __shared__ __bf16 Vt[64 * 64];      // [d][kpos]  (transposed V for P*V)
  __shared__ float  Ss[4][16 * 64];   // per-wave raw scores
  __shared__ __bf16 Ps[4][16 * 64];   // per-wave softmaxed probs (bf16)

  const int n = blockIdx.x;    // head within axis group
  const int p = blockIdx.y;    // fixed coordinate (w for AXIS 0, h for AXIS 1)
  const int b = blockIdx.z;

  size_t base; size_t rstride;
  if (AXIS == 0) { base = ((size_t)(b * 64 * 64) + p) * 768 + (size_t)n * 64; rstride = (size_t)64 * 768; }
  else           { base = ((size_t)(b * 64 + p) * 64) * 768 + (size_t)(6 + n) * 64; rstride = 768; }

  const int tid  = threadIdx.x;
  const int wv   = tid >> 5, lane = tid & 31;
  const int lh   = lane >> 4, lrow = lane & 15;
  const int qr   = wv * 16;

  // ---- stage Q, K (async pure copies) and V (manual transpose) ----
  {
    const int r = tid >> 1, c0 = (tid & 1) * 32;   // 32 bf16 per thread
    const __bf16* gq = q + base + (size_t)r * rstride + c0;
    const __bf16* gk = k + base + (size_t)r * rstride + c0;
    const __bf16* gv = v + base + (size_t)r * rstride + c0;
#pragma unroll
    for (int j = 0; j < 4; ++j) {
      async_copy_b128(lds_off32(&Qs[r * 64 + c0 + j * 8]), gq + j * 8);
      async_copy_b128(lds_off32(&Ks[r * 64 + c0 + j * 8]), gk + j * 8);
      union { uint4 u; __bf16 h[8]; } t;
      t.u = *(const uint4*)(gv + j * 8);
#pragma unroll
      for (int e = 0; e < 8; ++e) Vt[(c0 + j * 8 + e) * 64 + r] = t.h[e];
    }
  }
  async_wait_le<0>();
  __syncthreads();

  // ---- S = Q * K^T (16 q-rows x 64 k-pos per wave), K-dim 64 = 2 chunks ----
  v8f sacc[4] = {};
  {
    v16bf aq[2];
#pragma unroll
    for (int dc = 0; dc < 2; ++dc)
      aq[dc] = frag_a(&Qs[(qr + lrow) * 64 + dc * 32], lh);
#pragma unroll
    for (int kt = 0; kt < 4; ++kt)
#pragma unroll
      for (int dc = 0; dc < 2; ++dc) {
        v16bf bk = frag_b(&Ks[(kt * 16 + lrow) * 64 + dc * 32], lh);
        sacc[kt] = __builtin_amdgcn_wmma_f32_16x16x32_bf16(
            false, aq[dc], false, bk, (short)0, sacc[kt], false, false);
      }
  }
  // scatter scores (scaled by 1/sqrt(64)) to per-wave LDS
#pragma unroll
  for (int kt = 0; kt < 4; ++kt)
#pragma unroll
    for (int vv = 0; vv < 8; ++vv)
      Ss[wv][(lh * 8 + vv) * 64 + kt * 16 + lrow] = sacc[kt][vv] * 0.125f;
  __syncthreads();

  // ---- softmax: lanes 0..15 each own one full row of 64 scores ----
  if (lane < 16) {
    float* row = &Ss[wv][lane * 64];
    float mx = row[0];
#pragma unroll 8
    for (int i = 1; i < 64; ++i) mx = fmaxf(mx, row[i]);
    float sum = 0.f;
#pragma unroll 8
    for (int i = 0; i < 64; ++i) { float e = __expf(row[i] - mx); row[i] = e; sum += e; }
    const float inv = 1.f / sum;
#pragma unroll 8
    for (int i = 0; i < 64; ++i) Ps[wv][lane * 64 + i] = (__bf16)(row[i] * inv);
  }
  __syncthreads();

  // ---- O = P * V : A = P[16x64], B from Vt (lane holds d column) ----
  v8f oacc[4] = {};
  {
    v16bf ap[2];
#pragma unroll
    for (int xc = 0; xc < 2; ++xc)
      ap[xc] = frag_a(&Ps[wv][lrow * 64 + xc * 32], lh);
#pragma unroll
    for (int dt = 0; dt < 4; ++dt)
#pragma unroll
      for (int xc = 0; xc < 2; ++xc) {
        v16bf bv = frag_b(&Vt[(dt * 16 + lrow) * 64 + xc * 32], lh);
        oacc[dt] = __builtin_amdgcn_wmma_f32_16x16x32_bf16(
            false, ap[xc], false, bv, (short)0, oacc[dt], false, false);
      }
  }
  // ---- store O (bf16) back in [b,h,w,nh,d] layout ----
#pragma unroll
  for (int dt = 0; dt < 4; ++dt) {
    const int d = dt * 16 + lrow;
#pragma unroll
    for (int vv = 0; vv < 8; ++vv) {
      const int rowp = qr + lh * 8 + vv;
      o[base + (size_t)rowp * rstride + d] = (__bf16)oacc[dt][vv];
    }
  }
}

// ---------------------------------------------------------------------------
extern "C" void kernel_launch(void* const* d_in, const int* in_sizes, int n_in,
                              void* d_out, int out_size, void* d_ws, size_t ws_size,
                              hipStream_t stream) {
  const float* queries = (const float*)d_in[0];
  const float* values  = (const float*)d_in[1];
  const float* Wq = (const float*)d_in[2];
  const float* bq = (const float*)d_in[3];
  const float* Wk = (const float*)d_in[4];
  const float* bk = (const float*)d_in[5];
  const float* Wv = (const float*)d_in[6];
  const float* bv = (const float*)d_in[7];
  const float* Wo = (const float*)d_in[8];  // [NH,KD,C] == [768,768] row-major
  const float* bo = (const float*)d_in[9];

  char* ws = (char*)d_ws;
  const size_t mat  = (size_t)MTOT * NDIM * 2;   // one bf16 [65536,768] matrix
  const size_t wmat = (size_t)KDIM * NDIM * 2;   // one bf16 [768,768] weight
  void* qws = ws;
  void* kws = ws + mat;
  void* vws = ws + 2 * mat;
  void* ows = ws + 3 * mat;
  __bf16* wqT = (__bf16*)(ws + 4 * mat);
  __bf16* wkT = (__bf16*)(ws + 4 * mat + wmat);
  __bf16* wvT = (__bf16*)(ws + 4 * mat + 2 * wmat);
  __bf16* woT = (__bf16*)(ws + 4 * mat + 3 * wmat);

  // weight pre-transpose (tiny; stays resident in L2)
  dim3 gw(NDIM / 256, KDIM, 1), bw(256);
  transpose_w_bf16<<<gw, bw, 0, stream>>>(Wq, wqT);
  transpose_w_bf16<<<gw, bw, 0, stream>>>(Wk, wkT);
  transpose_w_bf16<<<gw, bw, 0, stream>>>(Wv, wvT);
  transpose_w_bf16<<<gw, bw, 0, stream>>>(Wo, woT);

  dim3 gp(NDIM / 128, MTOT / 128, 1), bp(256);
  fused_proj_gemm<false, false><<<gp, bp, 0, stream>>>(queries, wqT, bq, qws);
  fused_proj_gemm<false, false><<<gp, bp, 0, stream>>>(values,  wkT, bk, kws);
  fused_proj_gemm<false, false><<<gp, bp, 0, stream>>>(values,  wvT, bv, vws);

  dim3 ga(6, 64, 16), ba(128);
  axis_attn<0><<<ga, ba, 0, stream>>>((const __bf16*)qws, (const __bf16*)kws,
                                      (const __bf16*)vws, (__bf16*)ows);
  axis_attn<1><<<ga, ba, 0, stream>>>((const __bf16*)qws, (const __bf16*)kws,
                                      (const __bf16*)vws, (__bf16*)ows);

  fused_proj_gemm<true, true><<<gp, bp, 0, stream>>>(ows, woT, bo, d_out);
}
